// LSTM3DCell_65214783423147
// MI455X (gfx1250) — matrix operand order repaired and checked by reading, not compiled, not run on hardware
//
#include <hip/hip_runtime.h>
#include <hip/hip_bf16.h>

typedef __attribute__((ext_vector_type(16))) __bf16 v16bf;
typedef __attribute__((ext_vector_type(8)))  __bf16 v8bf;
typedef __attribute__((ext_vector_type(8)))  float  v8f;

#define B_   32
#define P_   512     // D*H*W
#define CIN_ 512
#define F_   96
#define NT_  6       // 96/16 N-tiles

static __device__ __forceinline__ v16bf join16(v8bf lo, v8bf hi) {
  return __builtin_shufflevector(lo, hi, 0,1,2,3,4,5,6,7,8,9,10,11,12,13,14,15);
}
static __device__ __forceinline__ v8bf zero8() {
  v8bf z;
  #pragma unroll
  for (int i = 0; i < 8; ++i) z[i] = (__bf16)0.0f;
  return z;
}
static __device__ __forceinline__ unsigned lds_off(const void* p) {
  // flat shared address: low 32 bits are the LDS byte offset
  return (unsigned)(unsigned long long)p;
}
static __device__ __forceinline__ void async_wait0() {
  asm volatile("s_wait_asynccnt 0x0" ::: "memory");
}

// ---------------------------------------------------------------------------
// Prep kernels: one-time f32 -> bf16 conversions of reused operands
// ---------------------------------------------------------------------------
__global__ __launch_bounds__(256) void lstm3d_prep_x(const float* __restrict__ X,
                                                     __bf16* __restrict__ Xb) {
  const int idx = blockIdx.x * 256 + threadIdx.x;           // 32*512
  Xb[idx] = (__bf16)X[idx];
}
__global__ __launch_bounds__(256) void lstm3d_prep_h(const float* __restrict__ H,
                                                     __bf16* __restrict__ Hb) {
  const int idx = blockIdx.x * 256 + threadIdx.x;           // 32*512*96
  Hb[idx] = (__bf16)H[idx];
}
__global__ __launch_bounds__(256) void lstm3d_prep_u(
    const float* __restrict__ Uf, const float* __restrict__ Ui,
    const float* __restrict__ Us, __bf16* __restrict__ Ub) {
  const int idx = blockIdx.x * 256 + threadIdx.x;           // 3*27*96*96
  const int c   = idx % F_;
  const int f   = (idx / F_) % F_;
  const int tap = (idx / (F_ * F_)) % 27;
  const int g   = idx / (F_ * F_ * 27);
  const float* Ug = (g == 0) ? Uf : (g == 1) ? Ui : Us;
  // Ub[g][tap][f][c]  <-  U_g[tap][c][f]   (pre-transposed for B-frag reads)
  Ub[idx] = (__bf16)Ug[(size_t)tap * (F_ * F_) + c * F_ + f];
}

// ---------------------------------------------------------------------------
// Kernel 1: G[g,b,p,f] = X @ W_g[p]  — async double-buffered weight streaming
// One wave per (position, gate); weights NT-streamed from HBM once.
// ---------------------------------------------------------------------------
__global__ __launch_bounds__(32) void lstm3d_einsum(
    const __bf16* __restrict__ Xb,
    const float* __restrict__ Wf, const float* __restrict__ Wi,
    const float* __restrict__ Ws, float* __restrict__ G)
{
  __shared__ float sW[2][32 * F_];           // two 12 KB chunk buffers
  const int g = blockIdx.x % 3;
  const int p = blockIdx.x / 3;
  const float* Wg = (g == 0) ? Wf : (g == 1) ? Wi : Ws;
  const float* Wp = Wg + (size_t)p * (CIN_ * F_);

  const int lane = threadIdx.x & 31;
  const int hf   = lane >> 4;
  const int l16  = lane & 15;

  v8f acc[2][NT_];
  #pragma unroll
  for (int m = 0; m < 2; ++m)
    #pragma unroll
    for (int n = 0; n < NT_; ++n)
      #pragma unroll
      for (int e = 0; e < 8; ++e) acc[m][n][e] = 0.0f;

  // Prologue: async-copy chunk 0 (contiguous 12 KB) into buffer 0.
  {
    const char* src = (const char*)Wp;
    const unsigned dst = lds_off(&sW[0][0]);
    #pragma unroll
    for (int j = 0; j < 24; ++j) {
      const unsigned off = (unsigned)(lane + j * 32) * 16u;
      asm volatile("global_load_async_to_lds_b128 %0, %1, off th:TH_LOAD_NT"
                   :: "v"(dst + off), "v"(src + off) : "memory");
    }
  }

  for (int t = 0; t < CIN_ / 32; ++t) {
    const int c0 = t * 32;
    async_wait0();                           // chunk t landed in LDS

    if (t + 1 < CIN_ / 32) {                 // overlap: start chunk t+1
      const char* src = (const char*)(Wp + (size_t)(c0 + 32) * F_);
      const unsigned dst = lds_off(&sW[(t + 1) & 1][0]);
      #pragma unroll
      for (int j = 0; j < 24; ++j) {
        const unsigned off = (unsigned)(lane + j * 32) * 16u;
        asm volatile("global_load_async_to_lds_b128 %0, %1, off th:TH_LOAD_NT"
                     :: "v"(dst + off), "v"(src + off) : "memory");
      }
    }

    // A fragments (bf16 X, L2-hot): two 16-B loads per m-tile.
    v16bf afrag[2];
    #pragma unroll
    for (int m = 0; m < 2; ++m) {
      const __bf16* xr = Xb + (size_t)(m * 16 + l16) * CIN_ + c0 + hf * 16;
      afrag[m] = join16(*(const v8bf*)xr, *(const v8bf*)(xr + 8));
    }

    const float* buf = sW[t & 1];
    #pragma unroll
    for (int n = 0; n < NT_; ++n) {
      const int f = n * 16 + l16;
      v16bf bfrag;
      #pragma unroll
      for (int j = 0; j < 16; ++j)
        bfrag[j] = (__bf16)buf[(hf * 16 + j) * F_ + f];
      acc[0][n] = __builtin_amdgcn_wmma_f32_16x16x32_bf16(
          false, afrag[0], false, bfrag, (short)0, acc[0][n], false, false);
      acc[1][n] = __builtin_amdgcn_wmma_f32_16x16x32_bf16(
          false, afrag[1], false, bfrag, (short)0, acc[1][n], false, false);
    }
  }

  #pragma unroll
  for (int m = 0; m < 2; ++m)
    #pragma unroll
    for (int n = 0; n < NT_; ++n)
      #pragma unroll
      for (int r = 0; r < 8; ++r) {
        const int b = m * 16 + hf * 8 + r;
        const size_t idx = (((size_t)g * B_ + b) * P_ + p) * F_ + n * 16 + l16;
        G[idx] = acc[m][n][r];
      }
}

// ---------------------------------------------------------------------------
// Kernel 2: G += conv3d_same(h_t, U_g) — async double-buffered tap slabs
// 8 waves/block share each 18 KB bf16 tap slab (pre-transposed [f][c]).
// ---------------------------------------------------------------------------
__global__ __launch_bounds__(256) void lstm3d_conv(
    const __bf16* __restrict__ Hb, const __bf16* __restrict__ Ub,
    float* __restrict__ G)
{
  __shared__ __bf16 sU[2][F_ * F_];          // two 18 KB tap buffers
  const int g    = blockIdx.x % 3;
  const int mb   = blockIdx.x / 3;           // 0..127
  const int tid  = threadIdx.x;
  const int wave = tid >> 5;
  const int lane = tid & 31;
  const int hf = lane >> 4, l16 = lane & 15;
  const int mtile = mb * 8 + wave;           // 0..1023

  const __bf16* Ug = Ub + (size_t)g * 27 * (F_ * F_);

  const int R = mtile * 16 + l16;            // this lane's A-row
  const int b = R >> 9;
  const int p = R & 511;
  const int d = p >> 6, hh = (p >> 3) & 7, ww = p & 7;

  v8f acc[NT_];
  #pragma unroll
  for (int n = 0; n < NT_; ++n)
    #pragma unroll
    for (int e = 0; e < 8; ++e) acc[n][e] = 0.0f;

  // Prologue: async-copy tap 0 slab (contiguous 18432 B) into buffer 0.
  {
    const char* src = (const char*)Ug;
    const unsigned dst = lds_off(&sU[0][0]);
    #pragma unroll
    for (int i = 0; i < 9; ++i) {
      const unsigned off = (unsigned)(tid + i * 256) * 8u;
      asm volatile("global_load_async_to_lds_b64 %0, %1, off"
                   :: "v"(dst + off), "v"(src + off) : "memory");
    }
  }

  for (int tap = 0; tap < 27; ++tap) {
    async_wait0();                            // my wave's copies done
    __syncthreads();                          // everyone's copies visible

    if (tap + 1 < 27) {                       // overlap next tap slab
      const char* src = (const char*)(Ug + (size_t)(tap + 1) * (F_ * F_));
      const unsigned dst = lds_off(&sU[(tap + 1) & 1][0]);
      #pragma unroll
      for (int i = 0; i < 9; ++i) {
        const unsigned off = (unsigned)(tid + i * 256) * 8u;
        asm volatile("global_load_async_to_lds_b64 %0, %1, off"
                     :: "v"(dst + off), "v"(src + off) : "memory");
      }
    }

    const int tz = tap / 9 - 1;
    const int ty = (tap / 3) % 3 - 1;
    const int tx = tap % 3 - 1;
    const int nd = d + tz, nh = hh + ty, nw = ww + tx;
    const bool valid = ((unsigned)nd < 8u) & ((unsigned)nh < 8u) & ((unsigned)nw < 8u);
    const __bf16* hrow = Hb + (size_t)(b * 512 + ((nd << 6) + (nh << 3) + nw)) * F_;
    const __bf16* ub = sU[tap & 1];

    #pragma unroll
    for (int kc = 0; kc < 3; ++kc) {
      const int cbase = kc * 32 + hf * 16;
      v8bf alo = zero8(), ahi = zero8();
      if (valid) {                            // SAME padding -> zeros
        alo = *(const v8bf*)(hrow + cbase);
        ahi = *(const v8bf*)(hrow + cbase + 8);
      }
      const v16bf a = join16(alo, ahi);

      #pragma unroll
      for (int n = 0; n < NT_; ++n) {
        const __bf16* bp = &ub[(n * 16 + l16) * F_ + kc * 32 + hf * 16];
        const v16bf bfrag = join16(*(const v8bf*)bp, *(const v8bf*)(bp + 8));
        acc[n] = __builtin_amdgcn_wmma_f32_16x16x32_bf16(
            false, a, false, bfrag, (short)0, acc[n], false, false);
      }
    }
    __syncthreads();                          // reads done before next overwrite
  }

  #pragma unroll
  for (int n = 0; n < NT_; ++n)
    #pragma unroll
    for (int r = 0; r < 8; ++r) {
      const int Rr = mtile * 16 + hf * 8 + r;
      const int bb = Rr >> 9, pp = Rr & 511;
      const size_t idx = (((size_t)g * B_ + bb) * P_ + pp) * F_ + n * 16 + l16;
      G[idx] += acc[n][r];
    }
}

// ---------------------------------------------------------------------------
// Kernel 3: pointwise gate math + state update
// ---------------------------------------------------------------------------
__global__ __launch_bounds__(256) void lstm3d_pointwise(
    const float* __restrict__ G, const float* __restrict__ s_t,
    const float* __restrict__ bf_, const float* __restrict__ bi_,
    const float* __restrict__ bs_, float* __restrict__ out)
{
  const size_t S = (size_t)B_ * P_ * F_;
  const size_t idx = (size_t)blockIdx.x * blockDim.x + threadIdx.x;
  if (idx >= S) return;
  const int f = (int)(idx % F_);
  const float gf = G[idx]         + bf_[f];
  const float gi = G[S + idx]     + bi_[f];
  const float gs = G[2 * S + idx] + bs_[f];
  const float fg = fminf(fmaxf(0.2f * gf + 0.5f, 0.0f), 1.0f);  // hard_sigmoid
  const float ig = fminf(fmaxf(0.2f * gi + 0.5f, 0.0f), 1.0f);
  const float s  = fg * s_t[idx] + ig * tanhf(gs);
  out[idx]     = tanhf(s);   // h
  out[S + idx] = s;          // s
}

extern "C" void kernel_launch(void* const* d_in, const int* in_sizes, int n_in,
                              void* d_out, int out_size, void* d_ws, size_t ws_size,
                              hipStream_t stream) {
  const float* X  = (const float*)d_in[0];
  const float* Ht = (const float*)d_in[1];
  const float* St = (const float*)d_in[2];
  const float* Wf = (const float*)d_in[3];
  const float* Wi = (const float*)d_in[4];
  const float* Ws = (const float*)d_in[5];
  const float* Uf = (const float*)d_in[6];
  const float* Ui = (const float*)d_in[7];
  const float* Us = (const float*)d_in[8];
  const float* bf = (const float*)d_in[9];
  const float* bi = (const float*)d_in[10];
  const float* bs = (const float*)d_in[11];

  char* wsB = (char*)d_ws;
  float*  G  = (float*)wsB;                                   // 18,874,368 B
  __bf16* Xb = (__bf16*)(wsB + 18874368);                     //     32,768 B
  __bf16* Hb = (__bf16*)(wsB + 18874368 + 32768);             //  3,145,728 B
  __bf16* Ub = (__bf16*)(wsB + 18874368 + 32768 + 3145728);   //  1,492,992 B
  float* out = (float*)d_out;

  const int S = B_ * P_ * F_;
  lstm3d_prep_x   <<<(B_ * CIN_) / 256, 256, 0, stream>>>(X, Xb);
  lstm3d_prep_h   <<<S / 256, 256, 0, stream>>>(Ht, Hb);
  lstm3d_prep_u   <<<(3 * 27 * F_ * F_) / 256, 256, 0, stream>>>(Uf, Ui, Us, Ub);
  lstm3d_einsum   <<<P_ * 3, 32,  0, stream>>>(Xb, Wf, Wi, Ws, G);
  lstm3d_conv     <<<128 * 3, 256, 0, stream>>>(Hb, Ub, G);
  lstm3d_pointwise<<<(S + 255) / 256, 256, 0, stream>>>(G, St, bf, bi, bs, out);
}
